// GraphAttentionEmbedding_30966714204565
// MI455X (gfx1250) — compile-verified, matrix operand order checked by reading, not run on hardware
//
#include <hip/hip_runtime.h>
#include <math.h>

#define N_NODES 50000
#define N_EDGES 200000
#define IN_CH   256
#define TDIM_   100
#define MSG_DIM 172
#define EDIM_   272
#define KPAD    288   // EDIM padded to multiple of 32
#define HEADS   2
#define CH      128   // per-head channels
#define NACC    8     // 16x128 strip per wave

typedef __attribute__((ext_vector_type(16))) __bf16 v16bf;
typedef __attribute__((ext_vector_type(8)))  __bf16 v8bf;
typedef __attribute__((ext_vector_type(8)))  float  v8f;

// ---------------------------------------------------------------------------
// atomic float max via int/uint ordering trick (init value must be -inf)
__device__ __forceinline__ void atomic_max_f32(float* addr, float val) {
    if (val >= 0.0f) {
        atomicMax((int*)addr, __float_as_int(val));
    } else {
        atomicMin((unsigned int*)addr, (unsigned int)__float_as_int(val));
    }
}

// ---------------------------------------------------------------------------
// x (f32) -> bf16
__global__ void k_cvt_x(const float* __restrict__ x, __bf16* __restrict__ xbf, int n) {
    int i = blockIdx.x * blockDim.x + threadIdx.x;
    if (i < n) xbf[i] = (__bf16)x[i];
}

// bias1024 = [bq | bk | bv | bskip]
__global__ void k_bias(const float* bq, const float* bk, const float* bv,
                       const float* bs, float* __restrict__ bias) {
    int i = blockIdx.x * blockDim.x + threadIdx.x;
    if (i >= 1024) return;
    int g = i >> 8, c = i & 255;
    const float* src = (g == 0) ? bq : (g == 1) ? bk : (g == 2) ? bv : bs;
    bias[i] = src[c];
}

// amax = -inf, denom = 0
__global__ void k_init_nm(float* __restrict__ amax, float* __restrict__ denom, int n) {
    int i = blockIdx.x * blockDim.x + threadIdx.x;
    if (i < n) { amax[i] = -INFINITY; denom[i] = 0.0f; }
}

// Swizzle Wcat = [Wq|Wk|Wv|Wskip] (K=256, Ncols=1024) into WMMA-B fragment
// layout: tile (kt,nt) -> 512 bf16 elems, lane-major 16 elems per lane.
// Per lane: n = nt*16 + (lane&15); klo = (lane&16)?8:0; elem i -> k = kt*32+klo+(i<8?i:i+8)
__global__ void k_swz_wcat(const float* Wq, const float* Wk, const float* Wv,
                           const float* Ws, __bf16* __restrict__ bsw) {
    int tid = blockIdx.x * blockDim.x + threadIdx.x;
    const int total = 8 * 64 * 512;             // kTiles * nTiles * 512
    if (tid >= total) return;
    int tile = tid >> 9, r = tid & 511;
    int lane = r >> 4, i = r & 15;
    int kt = tile >> 6, nt = tile & 63;         // nc16 = 64
    int klo = (lane & 16) ? 8 : 0;
    int k = kt * 32 + klo + (i < 8 ? i : i + 8);
    int n = nt * 16 + (lane & 15);
    int g = n >> 8, c = n & 255;
    const float* W = (g == 0) ? Wq : (g == 1) ? Wk : (g == 2) ? Wv : Ws;
    bsw[tid] = (__bf16)W[k * 256 + c];
}

// Swizzle We (K=272 padded to 288, Ncols=256) same layout; pad rows are 0.
__global__ void k_swz_we(const float* We, __bf16* __restrict__ bsw) {
    int tid = blockIdx.x * blockDim.x + threadIdx.x;
    const int total = 9 * 16 * 512;             // kTiles=9, nc16=16
    if (tid >= total) return;
    int tile = tid >> 9, r = tid & 511;
    int lane = r >> 4, i = r & 15;
    int kt = tile >> 4, nt = tile & 15;
    int klo = (lane & 16) ? 8 : 0;
    int k = kt * 32 + klo + (i < 8 ? i : i + 8);
    int n = nt * 16 + (lane & 15);
    float v = (k < EDIM_) ? We[k * 256 + n] : 0.0f;
    bsw[tid] = (__bf16)v;
}

// edge_attr[e, 0:100]   = cos((last_update[src]-t)*time_w + time_b)
// edge_attr[e, 100:272] = msg ; [272:288] = 0   (bf16, row stride 288)
__global__ void k_edge_attr(const int* __restrict__ ei, const float* __restrict__ lu,
                            const float* __restrict__ t, const float* __restrict__ msg,
                            const float* __restrict__ tw, const float* __restrict__ tb,
                            __bf16* __restrict__ ea) {
    long long tid = (long long)blockIdx.x * blockDim.x + threadIdx.x;
    const long long total = (long long)N_EDGES * KPAD;
    if (tid >= total) return;
    int edge = (int)(tid / KPAD);
    int col  = (int)(tid % KPAD);
    float v;
    if (col < TDIM_) {
        int src = ei[edge];
        float rel = lu[src] - t[edge];
        v = cosf(rel * tw[col] + tb[col]);
    } else if (col < EDIM_) {
        v = msg[(long long)edge * MSG_DIM + (col - TDIM_)];
    } else {
        v = 0.0f;
    }
    ea[tid] = (__bf16)v;
}

// ---------------------------------------------------------------------------
// bf16 WMMA GEMM: C[M, nc16*16] = A[M, kTiles*32] @ Bsw + bias
// One wave computes a 16x128 strip (NACC=8 accumulators) -> each A fragment
// feeds 8 WMMAs, halving A re-read traffic vs a 16x64 strip.
// Columns < ncols0 go to out0 (ld0), the rest to out1 (ld1), shifted.
__global__ void k_gemm_bf16(const __bf16* __restrict__ A, int lda, int mTiles,
                            const __bf16* __restrict__ Bsw, int kTiles, int nc16,
                            const float* __restrict__ bias,
                            float* __restrict__ out0, int ld0, int ncols0,
                            float* __restrict__ out1, int ld1,
                            int nColGroups) {
    int wave = (blockIdx.x * blockDim.x + threadIdx.x) >> 5;
    int lane = threadIdx.x & 31;
    if (wave >= mTiles * nColGroups) return;
    int rt = wave / nColGroups;
    int cg = wave % nColGroups;

    int row = rt * 16 + (lane & 15);
    int klo = (lane & 16) ? 8 : 0;
    const __bf16* arow = A + (size_t)row * lda + klo;

    v8f acc[NACC];
#pragma unroll
    for (int u = 0; u < NACC; ++u)
        acc[u] = (v8f){0.f, 0.f, 0.f, 0.f, 0.f, 0.f, 0.f, 0.f};

    for (int kt = 0; kt < kTiles; ++kt) {
        const __bf16* ap = arow + kt * 32;
        v8bf alo = *(const v8bf*)(ap);        // k = klo .. klo+7
        v8bf ahi = *(const v8bf*)(ap + 16);   // k = klo+16 .. klo+23
        v16bf a;
#pragma unroll
        for (int i = 0; i < 8; ++i) { a[i] = alo[i]; a[i + 8] = ahi[i]; }

        const __bf16* bbase = Bsw + ((size_t)kt * nc16 + cg * NACC) * 512 + lane * 16;
#pragma unroll
        for (int u = 0; u < NACC; ++u) {
            v8bf blo = *(const v8bf*)(bbase + (size_t)u * 512);
            v8bf bhi = *(const v8bf*)(bbase + (size_t)u * 512 + 8);
            v16bf b;
#pragma unroll
            for (int i = 0; i < 8; ++i) { b[i] = blo[i]; b[i + 8] = bhi[i]; }
            acc[u] = __builtin_amdgcn_wmma_f32_16x16x32_bf16(
                false, a, false, b, (short)0, acc[u], false, false);
        }
    }

    // C/D layout: lanes 0-15: N=lane, M=r; lanes 16-31: N=lane-16, M=r+8
    int mbase = rt * 16 + ((lane & 16) ? 8 : 0);
#pragma unroll
    for (int u = 0; u < NACC; ++u) {
        int col = (cg * NACC + u) * 16 + (lane & 15);
        float badd = bias ? bias[col] : 0.0f;
        float* dst; int ld, cc;
        if (col < ncols0) { dst = out0; ld = ld0; cc = col; }
        else              { dst = out1; ld = ld1; cc = col - ncols0; }
#pragma unroll
        for (int r = 0; r < 8; ++r)
            dst[(size_t)(mbase + r) * ld + cc] = acc[u][r] + badd;
    }
}

// ---------------------------------------------------------------------------
// One wave per (edge, head): alpha = dot(q[dst], k[src]+e) / sqrt(C)
__global__ void k_logits(const int* __restrict__ ei, const float* __restrict__ qkv,
                         const float* __restrict__ e, float* __restrict__ alpha,
                         float* __restrict__ amax) {
    int wave = (blockIdx.x * blockDim.x + threadIdx.x) >> 5;
    int lane = threadIdx.x & 31;
    if (wave >= N_EDGES * HEADS) return;
    int edge = wave >> 1, h = wave & 1;
    int src = ei[edge], dst = ei[N_EDGES + edge];

    const float4* qp = (const float4*)(qkv + (size_t)dst * 768 + h * CH);
    const float4* kp = (const float4*)(qkv + (size_t)src * 768 + 256 + h * CH);
    const float4* ep = (const float4*)(e + (size_t)edge * 256 + h * CH);
    float4 q4 = qp[lane], k4 = kp[lane], e4 = ep[lane];
    float d = q4.x * (k4.x + e4.x) + q4.y * (k4.y + e4.y) +
              q4.z * (k4.z + e4.z) + q4.w * (k4.w + e4.w);
#pragma unroll
    for (int off = 16; off > 0; off >>= 1) d += __shfl_xor(d, off, 32);
    if (lane == 0) {
        float a = d * 0.088388347648318447f;  // 1/sqrt(128)
        alpha[(size_t)edge * 2 + h] = a;
        atomic_max_f32(&amax[(size_t)dst * 2 + h], a);
    }
}

// ex = exp(alpha - amax[dst]); alpha <- ex; denom[dst] += ex
__global__ void k_expdenom(const int* __restrict__ ei, float* __restrict__ alpha,
                           const float* __restrict__ amax, float* __restrict__ denom) {
    int tid = blockIdx.x * blockDim.x + threadIdx.x;
    if (tid >= N_EDGES * HEADS) return;
    int edge = tid >> 1, h = tid & 1;
    int dst = ei[N_EDGES + edge];
    float ex = __expf(alpha[tid] - amax[(size_t)dst * 2 + h]);
    alpha[tid] = ex;
    atomicAdd(&denom[(size_t)dst * 2 + h], ex);
}

// One wave per edge: out[dst] += (v[src]+e) * (ex/denom[dst])  (256 floats)
__global__ void k_aggregate(const int* __restrict__ ei, const float* __restrict__ qkv,
                            const float* __restrict__ e, const float* __restrict__ ex,
                            const float* __restrict__ denom, float* __restrict__ out) {
    int wave = (blockIdx.x * blockDim.x + threadIdx.x) >> 5;
    int lane = threadIdx.x & 31;
    if (wave >= N_EDGES) return;
    int edge = wave;
    int src = ei[edge], dst = ei[N_EDGES + edge];
#pragma unroll
    for (int j = 0; j < 8; ++j) {
        int elem = j * 32 + lane;
        int h = elem >> 7;
        float w = ex[(size_t)edge * 2 + h] / denom[(size_t)dst * 2 + h];
        float val = (qkv[(size_t)src * 768 + 512 + elem] + e[(size_t)edge * 256 + elem]) * w;
        atomicAdd(&out[(size_t)dst * 256 + elem], val);
    }
}

// ---------------------------------------------------------------------------
extern "C" void kernel_launch(void* const* d_in, const int* in_sizes, int n_in,
                              void* d_out, int out_size, void* d_ws, size_t ws_size,
                              hipStream_t stream) {
    (void)in_sizes; (void)n_in; (void)out_size; (void)ws_size;
    const float* x      = (const float*)d_in[0];
    const float* lu     = (const float*)d_in[1];
    const int*   ei     = (const int*)  d_in[2];
    const float* t      = (const float*)d_in[3];
    const float* msg    = (const float*)d_in[4];
    const float* tw     = (const float*)d_in[5];
    const float* tb     = (const float*)d_in[6];
    const float* Wq     = (const float*)d_in[7];
    const float* bq     = (const float*)d_in[8];
    const float* Wk     = (const float*)d_in[9];
    const float* bk     = (const float*)d_in[10];
    const float* Wv     = (const float*)d_in[11];
    const float* bv     = (const float*)d_in[12];
    const float* We     = (const float*)d_in[13];
    const float* Ws     = (const float*)d_in[14];
    const float* bs     = (const float*)d_in[15];
    float* out = (float*)d_out;

    char* ws = (char*)d_ws;
    size_t off = 0;
    auto take = [&](size_t bytes) -> char* {
        char* p = ws + off;
        off = (off + bytes + 255) & ~(size_t)255;
        return p;
    };
    __bf16* xbf   = (__bf16*)take((size_t)N_NODES * IN_CH * 2);
    __bf16* wcat  = (__bf16*)take((size_t)8 * 64 * 512 * 2);
    __bf16* wep   = (__bf16*)take((size_t)9 * 16 * 512 * 2);
    float*  bias  = (float*) take(1024 * 4);
    float*  qkv   = (float*) take((size_t)N_NODES * 768 * 4);
    __bf16* ea    = (__bf16*)take((size_t)N_EDGES * KPAD * 2);
    float*  ebuf  = (float*) take((size_t)N_EDGES * 256 * 4);
    float*  alpha = (float*) take((size_t)N_EDGES * HEADS * 4);
    float*  amax  = (float*) take((size_t)N_NODES * HEADS * 4);
    float*  denom = (float*) take((size_t)N_NODES * HEADS * 4);

    const int TPB = 256;

    // --- prep ---
    k_cvt_x<<<(N_NODES * IN_CH + TPB - 1) / TPB, TPB, 0, stream>>>(x, xbf, N_NODES * IN_CH);
    k_bias<<<4, TPB, 0, stream>>>(bq, bk, bv, bs, bias);
    k_init_nm<<<(N_NODES * HEADS + TPB - 1) / TPB, TPB, 0, stream>>>(amax, denom, N_NODES * HEADS);
    k_swz_wcat<<<(8 * 64 * 512) / TPB, TPB, 0, stream>>>(Wq, Wk, Wv, Ws, wcat);
    k_swz_we<<<(9 * 16 * 512) / TPB, TPB, 0, stream>>>(We, wep);
    {
        long long total = (long long)N_EDGES * KPAD;
        int blocks = (int)((total + TPB - 1) / TPB);
        k_edge_attr<<<blocks, TPB, 0, stream>>>(ei, lu, t, msg, tw, tb, ea);
    }

    // --- GEMM1: [N,256] @ [256,1024] -> qkv (cols 0..767) + out(skip, cols 768..1023)
    {
        int mTiles = N_NODES / 16, nColGroups = 1024 / (16 * NACC);
        int waves = mTiles * nColGroups;
        k_gemm_bf16<<<(waves * 32 + TPB - 1) / TPB, TPB, 0, stream>>>(
            xbf, IN_CH, mTiles, wcat, 8, 64, bias,
            qkv, 768, 768, out, 256, nColGroups);
    }
    // --- GEMM2: [E,288] @ [288,256] -> ebuf
    {
        int mTiles = N_EDGES / 16, nColGroups = 256 / (16 * NACC);
        int waves = mTiles * nColGroups;
        k_gemm_bf16<<<(waves * 32 + TPB - 1) / TPB, TPB, 0, stream>>>(
            ea, KPAD, mTiles, wep, 9, 16, (const float*)nullptr,
            ebuf, 256, 256, ebuf, 256, nColGroups);
    }

    // --- attention ---
    {
        int waves = N_EDGES * HEADS;
        k_logits<<<(waves * 32 + TPB - 1) / TPB, TPB, 0, stream>>>(ei, qkv, ebuf, alpha, amax);
        k_expdenom<<<(N_EDGES * HEADS + TPB - 1) / TPB, TPB, 0, stream>>>(ei, alpha, amax, denom);
        k_aggregate<<<(N_EDGES * 32 + TPB - 1) / TPB, TPB, 0, stream>>>(ei, qkv, ebuf, alpha, denom, out);
    }
}
